// GemmaAttention_1108101562584
// MI455X (gfx1250) — compile-verified, matrix operand order checked by reading, not run on hardware
//
#include <hip/hip_runtime.h>

#define S_LEN 4096
#define HID   2048
#define NHEAD 8
#define HD    256
#define NB    2
#define NEGINF (-1.0e9f)

#define BLK_M 128
#define BLK_N 64
#define BK    32

typedef __attribute__((ext_vector_type(16))) __bf16 bf16x16;
typedef __attribute__((ext_vector_type(8)))  float  f32x8;

// ---------------- fragment loaders (16x16x32 bf16 WMMA layout) ----------------
// lane<16 -> K {0..7,16..23}, lane>=16 -> K {8..15,24..31}, row = row0 + (lane&15).
__device__ __forceinline__ bf16x16 frag_bf16(const __bf16* __restrict__ base,
                                             size_t ld, int row0, int k0, int lane) {
  const int half = lane >> 4;
  const int r    = lane & 15;
  const __bf16* p = base + (size_t)(row0 + r) * ld + (size_t)(k0 + (half << 3));
  union { uint4 q[2]; bf16x16 v; } u;
  u.q[0] = *(const uint4*)(p);
  u.q[1] = *(const uint4*)(p + 16);
  return u.v;
}

__device__ __forceinline__ bf16x16 frag_f32(const float* __restrict__ base,
                                            size_t ld, int row0, int k0, int lane) {
  const int half = lane >> 4;
  const int r    = lane & 15;
  const float* p = base + (size_t)(row0 + r) * ld + (size_t)(k0 + (half << 3));
  float4 f0 = *(const float4*)(p);
  float4 f1 = *(const float4*)(p + 4);
  float4 f2 = *(const float4*)(p + 16);
  float4 f3 = *(const float4*)(p + 20);
  bf16x16 v;
  v[0]=(__bf16)f0.x;  v[1]=(__bf16)f0.y;  v[2]=(__bf16)f0.z;  v[3]=(__bf16)f0.w;
  v[4]=(__bf16)f1.x;  v[5]=(__bf16)f1.y;  v[6]=(__bf16)f1.z;  v[7]=(__bf16)f1.w;
  v[8]=(__bf16)f2.x;  v[9]=(__bf16)f2.y;  v[10]=(__bf16)f2.z; v[11]=(__bf16)f2.w;
  v[12]=(__bf16)f3.x; v[13]=(__bf16)f3.y; v[14]=(__bf16)f3.z; v[15]=(__bf16)f3.w;
  return v;
}

// B fragment from LDS tile [BLK_N rows x BK cols] bf16, row-major.
__device__ __forceinline__ bf16x16 frag_lds(const __bf16* __restrict__ sm,
                                            int row0, int lane) {
  const int half = lane >> 4;
  const int r    = lane & 15;
  const __bf16* p = sm + (row0 + r) * BK + (half << 3);
  union { uint4 q[2]; bf16x16 v; } u;
  u.q[0] = *(const uint4*)(p);
  u.q[1] = *(const uint4*)(p + 16);
  return u.v;
}

// C/D layout: VGPR e -> M = e + 8*(lane>>4), N = n0 + (lane&15)
__device__ __forceinline__ void store_tile(void* __restrict__ Cv, int c_mode, size_t ldc,
                                           int m0, int n0, int lane, f32x8 c, bool skip) {
  const int half = lane >> 4;
  const int r    = lane & 15;
  const int n    = n0 + r;
  #pragma unroll
  for (int e = 0; e < 8; ++e) {
    const int m = m0 + (half << 3) + e;
    float val;
    if (skip) {
      val = NEGINF;
    } else {
      val = c[e];
      if (c_mode == 2) { val *= 0.0625f; if (n > m) val = NEGINF; }  // 1/sqrt(256), causal
    }
    if (c_mode == 1) ((__bf16*)Cv)[(size_t)m * ldc + n] = (__bf16)val;
    else             ((float*)Cv)[(size_t)m * ldc + n]  = val;
  }
}

// ---------------- TN WMMA GEMM, async-LDS-staged B, double-buffered ----------------
// C[m,n] = sum_k A[m,k] * Bt[n,k].  Block = 128(M) x 64(N), 8 waves, 32x32 per wave.
// c_mode: 0 = f32 out, 1 = bf16 out, 2 = scores (f32, scale + causal mask, block skip)
template <bool A32>
__global__ __launch_bounds__(256) void gemm_tn_wmma(
    const void* __restrict__ Av, const __bf16* __restrict__ Btv, void* __restrict__ Cv,
    int c_mode, size_t lda, size_t ldb, size_t ldc, int K, int heads,
    size_t sa_b, size_t sa_h, size_t sb_b, size_t sb_h, size_t sc_b, size_t sc_h)
{
  __shared__ __align__(16) __bf16 smB[2][BLK_N * BK];   // 2 x 4 KB

  const int z = blockIdx.z;
  const int b = z / heads, h = z - b * heads;
  const int tid = threadIdx.x, lane = tid & 31, wid = tid >> 5;
  const int wm = wid >> 1, wn = wid & 1;
  const int m_blk = blockIdx.y * BLK_M;
  const int n_blk = blockIdx.x * BLK_N;
  const int m0 = m_blk + wm * 32;
  const int n0 = n_blk + wn * 32;
  const int nloc = wn * 32;                  // wave's N offset inside the LDS tile

  const float*  Af  = (const float*)Av  + sa_b * (size_t)b + sa_h * (size_t)h;
  const __bf16* Abf = (const __bf16*)Av + sa_b * (size_t)b + sa_h * (size_t)h;
  const __bf16* Bt  = Btv + sb_b * (size_t)b + sb_h * (size_t)h;
  void* C;
  if (c_mode == 1) C = (void*)((__bf16*)Cv + sc_b * (size_t)b + sc_h * (size_t)h);
  else             C = (void*)((float*)Cv  + sc_b * (size_t)b + sc_h * (size_t)h);

  // Block entirely above the causal diagonal: uniform early-out (no barriers crossed).
  if (c_mode == 2 && n_blk > m_blk + (BLK_M - 1)) {
    const f32x8 dummy = {0.f,0.f,0.f,0.f,0.f,0.f,0.f,0.f};
    store_tile(C, c_mode, ldc, m0,      n0,      lane, dummy, true);
    store_tile(C, c_mode, ldc, m0,      n0 + 16, lane, dummy, true);
    store_tile(C, c_mode, ldc, m0 + 16, n0,      lane, dummy, true);
    store_tile(C, c_mode, ldc, m0 + 16, n0 + 16, lane, dummy, true);
    return;
  }

  // Async-copy mapping: 256 threads x 16B == 64x32 bf16 tile (one instr per wave per stage)
  const int crow   = tid >> 2;               // 0..63  (row within B tile)
  const int cchunk = tid & 3;                // 0..3   (16B chunk within row)
  const __bf16* gsrc = Bt + (size_t)(n_blk + crow) * ldb + (size_t)(cchunk << 3);
  const unsigned lds0 = (unsigned)(unsigned long long)(&smB[0][0])
                      + (unsigned)(crow * (BK * 2) + cchunk * 16);
  const unsigned lds1 = lds0 + (unsigned)(BLK_N * BK * 2);

  auto issue_stage = [&](int buf, int k) {
    const unsigned long long ga = (unsigned long long)(gsrc + k);
    const unsigned la = buf ? lds1 : lds0;
    asm volatile("global_load_async_to_lds_b128 %0, %1, off"
                 :: "v"(la), "v"(ga) : "memory");
  };

  const f32x8 zero = {0.f,0.f,0.f,0.f,0.f,0.f,0.f,0.f};
  f32x8 c00 = zero, c01 = zero, c10 = zero, c11 = zero;

  issue_stage(0, 0);
  int bufsel = 0;
  for (int k = 0; k < K; k += BK) {
    const bool has_next = (k + BK) < K;
    if (has_next) {
      issue_stage(bufsel ^ 1, k + BK);
      asm volatile("s_wait_asynccnt 0x1" ::: "memory");  // this wave's current stage done
    } else {
      asm volatile("s_wait_asynccnt 0x0" ::: "memory");
    }
    __syncthreads();                                     // all waves' stores visible

    bf16x16 a0, a1;
    if (A32) {
      a0 = frag_f32(Af, lda, m0,      k, lane);
      a1 = frag_f32(Af, lda, m0 + 16, k, lane);
    } else {
      a0 = frag_bf16(Abf, lda, m0,      k, lane);
      a1 = frag_bf16(Abf, lda, m0 + 16, k, lane);
    }
    if (has_next) {                                      // A lookahead via global_prefetch_b8
      if (A32) __builtin_prefetch(Af  + (size_t)(m0 + (lane & 15)) * lda + (size_t)(k + BK), 0, 1);
      else     __builtin_prefetch(Abf + (size_t)(m0 + (lane & 15)) * lda + (size_t)(k + BK), 0, 1);
    }
    const __bf16* sm = smB[bufsel];
    bf16x16 b0 = frag_lds(sm, nloc,      lane);
    bf16x16 b1 = frag_lds(sm, nloc + 16, lane);

    c00 = __builtin_amdgcn_wmma_f32_16x16x32_bf16(false, a0, false, b0, (short)0, c00, false, false);
    c01 = __builtin_amdgcn_wmma_f32_16x16x32_bf16(false, a0, false, b1, (short)0, c01, false, false);
    c10 = __builtin_amdgcn_wmma_f32_16x16x32_bf16(false, a1, false, b0, (short)0, c10, false, false);
    c11 = __builtin_amdgcn_wmma_f32_16x16x32_bf16(false, a1, false, b1, (short)0, c11, false, false);

    __syncthreads();                                     // protect buffer about to be refilled
    bufsel ^= 1;
  }

  store_tile(C, c_mode, ldc, m0,      n0,      lane, c00, false);
  store_tile(C, c_mode, ldc, m0,      n0 + 16, lane, c01, false);
  store_tile(C, c_mode, ldc, m0 + 16, n0,      lane, c10, false);
  store_tile(C, c_mode, ldc, m0 + 16, n0 + 16, lane, c11, false);
}

// ---------------- elementwise / reshape kernels ----------------
__global__ void f32_to_bf16_kernel(const float* __restrict__ in, __bf16* __restrict__ out, size_t n) {
  size_t i = (size_t)blockIdx.x * 256 + threadIdx.x;
  if (i < n) out[i] = (__bf16)in[i];
}

// in: [K,N] f32 -> out: [N,K] bf16
__global__ void wtrans_f32_to_bf16(const float* __restrict__ in, __bf16* __restrict__ out,
                                   int K, int N) {
  size_t i = (size_t)blockIdx.x * 256 + threadIdx.x;
  if (i >= (size_t)K * N) return;
  int k = (int)(i % (size_t)K);
  int n = (int)(i / (size_t)K);
  out[i] = (__bf16)in[(size_t)k * N + n];
}

// v: [B*S, HD] bf16 -> vt: [B, HD, S] bf16
__global__ void transpose_v_kernel(const __bf16* __restrict__ v, __bf16* __restrict__ vt) {
  size_t i = (size_t)blockIdx.x * 256 + threadIdx.x;
  if (i >= (size_t)NB * HD * S_LEN) return;
  int t = (int)(i % S_LEN);
  size_t rest = i / S_LEN;
  int d = (int)(rest % HD);
  int b = (int)(rest / HD);
  vt[i] = v[((size_t)b * S_LEN + t) * HD + d];
}

// RoPE in place on Q [B*S, NHEAD*HD]; pair (j, j+128) within each head, pos = s.
__global__ void rope_q_kernel(__bf16* __restrict__ q) {
  size_t i = (size_t)blockIdx.x * 256 + threadIdx.x;
  if (i >= (size_t)NB * S_LEN * NHEAD * 128) return;
  int j = (int)(i & 127);
  int h = (int)((i >> 7) & (NHEAD - 1));
  size_t row = i >> 10;                    // b*S + s
  int pos = (int)(row & (S_LEN - 1));
  float invf = __powf(10000.0f, -(float)(2 * j) * (1.0f / 256.0f));
  float ang = (float)pos * invf;
  float sn, cs;
  __sincosf(ang, &sn, &cs);
  __bf16* p = q + row * (size_t)HID + (size_t)h * HD + j;
  float x1 = (float)p[0], x2 = (float)p[128];
  p[0]   = (__bf16)(x1 * cs - x2 * sn);
  p[128] = (__bf16)(x2 * cs + x1 * sn);
}

// RoPE in place on K [B*S, HD]
__global__ void rope_k_kernel(__bf16* __restrict__ k) {
  size_t i = (size_t)blockIdx.x * 256 + threadIdx.x;
  if (i >= (size_t)NB * S_LEN * 128) return;
  int j = (int)(i & 127);
  size_t row = i >> 7;                     // b*S + t
  int pos = (int)(row & (S_LEN - 1));
  float invf = __powf(10000.0f, -(float)(2 * j) * (1.0f / 256.0f));
  float ang = (float)pos * invf;
  float sn, cs;
  __sincosf(ang, &sn, &cs);
  __bf16* p = k + row * (size_t)HD + j;
  float x1 = (float)p[0], x2 = (float)p[128];
  p[0]   = (__bf16)(x1 * cs - x2 * sn);
  p[128] = (__bf16)(x2 * cs + x1 * sn);
}

// Row softmax in place over rows of length S_LEN; one 256-thread block per row.
__global__ __launch_bounds__(256) void softmax_rows(float* __restrict__ attn) {
  const size_t row = blockIdx.x;
  float* p = attn + row * (size_t)S_LEN;
  __shared__ float red[8];
  const int tid = threadIdx.x, lane = tid & 31, wid = tid >> 5;

  float mx = -3.4e38f;
  for (int i = tid; i < S_LEN; i += 256) mx = fmaxf(mx, p[i]);
  for (int o = 16; o > 0; o >>= 1) mx = fmaxf(mx, __shfl_xor(mx, o, 32));
  if (lane == 0) red[wid] = mx;
  __syncthreads();
  float m = red[0];
  #pragma unroll
  for (int i = 1; i < 8; ++i) m = fmaxf(m, red[i]);
  __syncthreads();

  float sum = 0.f;
  for (int i = tid; i < S_LEN; i += 256) {
    float e = __expf(p[i] - m);
    p[i] = e;
    sum += e;
  }
  for (int o = 16; o > 0; o >>= 1) sum += __shfl_xor(sum, o, 32);
  if (lane == 0) red[wid] = sum;
  __syncthreads();
  float tot = 0.f;
  #pragma unroll
  for (int i = 0; i < 8; ++i) tot += red[i];
  const float inv = 1.0f / tot;
  for (int i = tid; i < S_LEN; i += 256) p[i] *= inv;
}

// ---------------- launch ----------------
extern "C" void kernel_launch(void* const* d_in, const int* in_sizes, int n_in,
                              void* d_out, int out_size, void* d_ws, size_t ws_size,
                              hipStream_t stream) {
  (void)in_sizes; (void)n_in; (void)out_size; (void)ws_size;
  const float* hidden = (const float*)d_in[0];
  // d_in[1] attention_mask (causal, synthesized analytically), d_in[2] position_ids (arange)
  const float* Wq = (const float*)d_in[3];
  const float* Wk = (const float*)d_in[4];
  const float* Wv = (const float*)d_in[5];
  const float* Wo = (const float*)d_in[6];

  float* out_proj = (float*)d_out;                                   // [B,S,H] f32
  float* out_attn = (float*)d_out + (size_t)NB * S_LEN * HID;        // [B,NH,S,S] f32

  char* ws = (char*)d_ws;
  size_t off = 0;
  auto alloc = [&](size_t bytes) -> void* {
    void* p = ws + off;
    off += (bytes + 255) & ~(size_t)255;
    return p;
  };
  __bf16* hb   = (__bf16*)alloc((size_t)NB * S_LEN * HID * 2);   // hidden bf16
  __bf16* wqt  = (__bf16*)alloc((size_t)HID * HID * 2);          // Wq^T [N,K]
  __bf16* wkt  = (__bf16*)alloc((size_t)HD * HID * 2);           // Wk^T
  __bf16* wvt  = (__bf16*)alloc((size_t)HD * HID * 2);           // Wv^T
  __bf16* wot  = (__bf16*)alloc((size_t)HID * HID * 2);          // Wo^T
  __bf16* qbuf = (__bf16*)alloc((size_t)NB * S_LEN * HID * 2);   // Q [B*S, 8*256]
  __bf16* kbuf = (__bf16*)alloc((size_t)NB * S_LEN * HD * 2);    // K [B*S, 256]
  __bf16* vbuf = (__bf16*)alloc((size_t)NB * S_LEN * HD * 2);    // V [B*S, 256]
  __bf16* vt   = (__bf16*)alloc((size_t)NB * HD * S_LEN * 2);    // V^T [B, 256, S]
  __bf16* ctx  = (__bf16*)alloc((size_t)NB * S_LEN * HID * 2);   // attn@V [B*S, 8*256]

  const dim3 blk(256, 1, 1);
  const int M = NB * S_LEN;  // 8192

  // Convert inputs / transpose weights to bf16 [N,K]
  f32_to_bf16_kernel<<<(unsigned)(((size_t)M * HID) / 256), blk, 0, stream>>>(hidden, hb, (size_t)M * HID);
  wtrans_f32_to_bf16<<<(unsigned)(((size_t)HID * HID) / 256), blk, 0, stream>>>(Wq, wqt, HID, HID);
  wtrans_f32_to_bf16<<<(unsigned)(((size_t)HID * HD) / 256), blk, 0, stream>>>(Wk, wkt, HID, HD);
  wtrans_f32_to_bf16<<<(unsigned)(((size_t)HID * HD) / 256), blk, 0, stream>>>(Wv, wvt, HID, HD);
  wtrans_f32_to_bf16<<<(unsigned)(((size_t)HID * HID) / 256), blk, 0, stream>>>(Wo, wot, HID, HID);

  // QKV projections (bf16 out)
  gemm_tn_wmma<false><<<dim3(HID / 64, M / 128, 1), blk, 0, stream>>>(
      hb, wqt, qbuf, 1, (size_t)HID, (size_t)HID, (size_t)HID, HID, 1, 0, 0, 0, 0, 0, 0);
  gemm_tn_wmma<false><<<dim3(HD / 64, M / 128, 1), blk, 0, stream>>>(
      hb, wkt, kbuf, 1, (size_t)HID, (size_t)HID, (size_t)HD, HID, 1, 0, 0, 0, 0, 0, 0);
  gemm_tn_wmma<false><<<dim3(HD / 64, M / 128, 1), blk, 0, stream>>>(
      hb, wvt, vbuf, 1, (size_t)HID, (size_t)HID, (size_t)HD, HID, 1, 0, 0, 0, 0, 0, 0);

  // RoPE + V transpose
  rope_q_kernel<<<(unsigned)(((size_t)NB * S_LEN * NHEAD * 128) / 256), blk, 0, stream>>>(qbuf);
  rope_k_kernel<<<(unsigned)(((size_t)NB * S_LEN * 128) / 256), blk, 0, stream>>>(kbuf);
  transpose_v_kernel<<<(unsigned)(((size_t)NB * HD * S_LEN) / 256), blk, 0, stream>>>(vbuf, vt);

  // Scores: per (b,h), A = Q slice [S,256] (ld HID), Bt = K [S,256] (already [t,d]),
  // C = attn f32 with scale + causal mask; above-diagonal blocks skipped.
  gemm_tn_wmma<false><<<dim3(S_LEN / 64, S_LEN / 128, NB * NHEAD), blk, 0, stream>>>(
      qbuf, kbuf, out_attn, 2, (size_t)HID, (size_t)HD, (size_t)S_LEN, HD, NHEAD,
      (size_t)S_LEN * HID, (size_t)HD,
      (size_t)S_LEN * HD, 0,
      (size_t)NHEAD * S_LEN * S_LEN, (size_t)S_LEN * S_LEN);

  // Softmax in place on the attn output region
  softmax_rows<<<(unsigned)(NB * NHEAD * S_LEN), blk, 0, stream>>>(out_attn);

  // attn @ V: A = attn f32 (converted to bf16 in fragment loader), Bt = V^T [d,t]
  gemm_tn_wmma<true><<<dim3(HD / 64, S_LEN / 128, NB * NHEAD), blk, 0, stream>>>(
      out_attn, vt, ctx, 1, (size_t)S_LEN, (size_t)S_LEN, (size_t)HID, S_LEN, NHEAD,
      (size_t)NHEAD * S_LEN * S_LEN, (size_t)S_LEN * S_LEN,
      (size_t)HD * S_LEN, 0,
      (size_t)S_LEN * HID, (size_t)HD);

  // Output projection (f32 out to d_out)
  gemm_tn_wmma<false><<<dim3(HID / 64, M / 128, 1), blk, 0, stream>>>(
      ctx, wot, out_proj, 0, (size_t)HID, (size_t)HID, (size_t)HID, HID, 1, 0, 0, 0, 0, 0, 0);
}
